// ScalableFOGNN_60215441489929
// MI455X (gfx1250) — matrix operand order repaired and checked by reading, not compiled, not run on hardware
//
#include <hip/hip_runtime.h>
#include <math.h>

// ---------------------------------------------------------------------------
// ScalableFOGNN forward for MI455X (gfx1250, wave32, WMMA).
// Dense GEMMs -> v_wmma_f32_16x16x32_bf16 (fp32 HBM, bf16 LDS tiles, f32 acc).
//   block tile 128x128, 8 waves (4Mx2N), wave tile 32x64 = 8 WMMA / K-step.
//   A LDS row-major [row][k], B LDS column-major [col][k] so every fragment
//   is two contiguous ds_load_b128. Interior tiles stage with float4 loads.
// Edge ops    -> L2-resident float atomics (dest matrices <= 41MB << 192MB L2).
// ---------------------------------------------------------------------------

#define HID  256
#define HEADS 4
#define CHD  64

typedef __attribute__((ext_vector_type(16))) __bf16 v16bf;
typedef __attribute__((ext_vector_type(8)))  float  v8f;

__device__ __forceinline__ unsigned short f2bf(float f) {
  unsigned u = __float_as_uint(f);
  unsigned r = (u + 0x7FFFu + ((u >> 16) & 1u)) >> 16;  // round-to-nearest-even
  return (unsigned short)r;
}
__device__ __forceinline__ float atomicMaxF(float* addr, float val) {
  int* ia = (int*)addr;
  int old = __float_as_int(*addr);
  while (val > __int_as_float(old)) {
    int assumed = old;
    old = atomicCAS(ia, assumed, __float_as_int(val));
    if (old == assumed) break;
  }
  return __int_as_float(old);
}

union Frag16 { uint4 q[2]; v16bf v; };
union Pack4  { unsigned short u[4]; uint2 v; };

// ---------------------------------------------------------------------------
// GEMM: C[M,N] = act( A[M,K] @ B[K,N] + bias + addsrc )
// act: 0 none, 1 relu, 2 elu
// ---------------------------------------------------------------------------
#define BM 128
#define BN 128
#define BK 32
#define LDS_T 40   // ushorts per row/col slot: 32 data + 8 pad (80B, 16B-aligned)

__global__ __launch_bounds__(256)
void gemm_bf16_wmma(const float* __restrict__ A, int lda,
                    const float* __restrict__ B, int ldb,
                    float* __restrict__ C, int ldc,
                    const float* __restrict__ bias,
                    const float* __restrict__ addsrc, int ldadd,
                    int M, int N, int K, int act)
{
  __shared__ unsigned short As[BM * LDS_T];  // [row][k]
  __shared__ unsigned short Bs[BN * LDS_T];  // [col][k]  (column-major tile)

  const int tid   = threadIdx.x;
  const int lane  = tid & 31;
  const int wave  = tid >> 5;
  const int waveM = wave >> 1;   // 0..3 -> 32 rows each
  const int waveN = wave & 1;    // 0..1 -> 64 cols each
  const int m0 = blockIdx.y * BM;
  const int n0 = blockIdx.x * BN;
  const int half = lane >> 4;    // lane 0-15 vs 16-31
  const int hrow = lane & 15;

  const bool mFull = (m0 + BM <= M);
  const bool nFull = (n0 + BN <= N);

  v8f acc[2][4];
  #pragma unroll
  for (int i = 0; i < 2; i++)
    #pragma unroll
    for (int j = 0; j < 4; j++)
      #pragma unroll
      for (int r = 0; r < 8; r++) acc[i][j][r] = 0.f;

  for (int k0 = 0; k0 < K; k0 += BK) {
    const bool kFull = (k0 + BK <= K);

    // ---- stage A tile (f32 -> bf16), [128 rows][32 k] ----
    if (mFull && kFull) {
      #pragma unroll
      for (int p = 0; p < 4; p++) {
        int id = tid + p * 256;            // 1024 float4 total
        int r = id >> 3, c = (id & 7) * 4;
        const float4 f = *(const float4*)&A[(long)(m0 + r) * lda + k0 + c];
        Pack4 pk;
        pk.u[0] = f2bf(f.x); pk.u[1] = f2bf(f.y);
        pk.u[2] = f2bf(f.z); pk.u[3] = f2bf(f.w);
        *(uint2*)&As[r * LDS_T + c] = pk.v;
      }
    } else {
      for (int i = tid; i < BM * BK; i += 256) {
        int r = i >> 5, c = i & 31;
        int gr = m0 + r, gc = k0 + c;
        float v = (gr < M && gc < K) ? A[(long)gr * lda + gc] : 0.f;
        As[r * LDS_T + c] = f2bf(v);
      }
    }
    // ---- stage B tile (f32 -> bf16), column-major [128 cols][32 k] ----
    if (nFull && kFull) {
      #pragma unroll
      for (int p = 0; p < 4; p++) {
        int id = tid + p * 256;            // 1024 float4 total
        int kk = id >> 5, c = (id & 31) * 4;
        const float4 f = *(const float4*)&B[(long)(k0 + kk) * ldb + n0 + c];
        Bs[(c + 0) * LDS_T + kk] = f2bf(f.x);
        Bs[(c + 1) * LDS_T + kk] = f2bf(f.y);
        Bs[(c + 2) * LDS_T + kk] = f2bf(f.z);
        Bs[(c + 3) * LDS_T + kk] = f2bf(f.w);
      }
    } else {
      for (int i = tid; i < BK * BN; i += 256) {
        int r = i >> 7, c = i & 127;
        int gr = k0 + r, gc = n0 + c;
        float v = (gr < K && gc < N) ? B[(long)gr * ldb + gc] : 0.f;
        Bs[c * LDS_T + r] = f2bf(v);
      }
    }
    // prefetch next K-tiles (global_prefetch_b8)
    if (k0 + BK < K) {
      int gr = m0 + (tid >> 1);
      if (gr < M) __builtin_prefetch(&A[(long)gr * lda + k0 + BK], 0, 0);
    }
    __syncthreads();

    // ---- fragments: each is two contiguous ds_load_b128 ----
    // A 16-bit 16x32 layout: lanes0-15 K=0..7|16..23, lanes16-31 K=8..15|24..31
    v16bf af[2];
    #pragma unroll
    for (int sm = 0; sm < 2; sm++) {
      const unsigned short* ap = &As[(waveM * 32 + sm * 16 + hrow) * LDS_T];
      Frag16 fa;
      fa.q[0] = *(const uint4*)(ap + half * 8);        // K = half*8 .. +7
      fa.q[1] = *(const uint4*)(ap + 16 + half * 8);   // K = 16+half*8 .. +7
      af[sm] = fa.v;
    }
    // B 16-bit 32x16 layout: lanes0-15 N=lane,K=0..7|16..23; lanes16-31 K=8..15|24..31
    #pragma unroll
    for (int sn = 0; sn < 4; sn++) {
      const unsigned short* bp = &Bs[(waveN * 64 + sn * 16 + hrow) * LDS_T];
      Frag16 fb;
      fb.q[0] = *(const uint4*)(bp + half * 8);        // K = half*8 .. +7
      fb.q[1] = *(const uint4*)(bp + 16 + half * 8);   // K = 16+half*8 .. +7
      #pragma unroll
      for (int sm = 0; sm < 2; sm++)
        acc[sm][sn] = __builtin_amdgcn_wmma_f32_16x16x32_bf16(
            false, af[sm], false, fb.v, (short)0, acc[sm][sn], false, false);
    }
    __syncthreads();
  }

  // ---- epilogue: bias + residual + activation ----
  #pragma unroll
  for (int sm = 0; sm < 2; sm++)
    #pragma unroll
    for (int sn = 0; sn < 4; sn++)
      #pragma unroll
      for (int r = 0; r < 8; r++) {
        int row = m0 + waveM * 32 + sm * 16 + half * 8 + r;
        int col = n0 + waveN * 64 + sn * 16 + hrow;
        if (row < M && col < N) {
          float v = acc[sm][sn][r];
          if (bias)   v += bias[col];
          if (addsrc) v += addsrc[(long)row * ldadd + col];
          if (act == 1)      v = v > 0.f ? v : 0.f;
          else if (act == 2) v = v > 0.f ? v : (expf(v) - 1.f);
          C[(long)row * ldc + col] = v;
        }
      }
}

// ---------------------------------------------------------------------------
// Elementwise / reduction kernels
// ---------------------------------------------------------------------------
__global__ void fill_f32(float* p, float v, long n) {
  long i = (long)blockIdx.x * 256 + threadIdx.x;
  if (i < n) p[i] = v;
}
__global__ void fill_rows(float* out, const float* __restrict__ rowv, long M, int N) {
  long i = (long)blockIdx.x * 256 + threadIdx.x;
  if (i < M * N) out[i] = rowv[i % N];
}
__global__ void concat2(const float* __restrict__ X, const float* __restrict__ Y,
                        float* out, long M, int N) {  // out[M,2N] = [X|Y]
  long i = (long)blockIdx.x * 256 + threadIdx.x;
  if (i >= M * 2 * N) return;
  long m = i / (2 * N); int c = (int)(i % (2 * N));
  out[i] = (c < N) ? X[m * N + c] : Y[m * N + (c - N)];
}
__global__ void bn_stats(const float* __restrict__ X, int M, int N,
                         float* mean, float* var) {
  __shared__ float s1[256], s2[256];
  int c = blockIdx.x;
  float a = 0.f, b = 0.f;
  for (int r = threadIdx.x; r < M; r += 256) {
    float v = X[(long)r * N + c];
    a += v; b += v * v;
  }
  s1[threadIdx.x] = a; s2[threadIdx.x] = b;
  __syncthreads();
  for (int s = 128; s > 0; s >>= 1) {
    if (threadIdx.x < s) { s1[threadIdx.x] += s1[threadIdx.x + s];
                           s2[threadIdx.x] += s2[threadIdx.x + s]; }
    __syncthreads();
  }
  if (threadIdx.x == 0) {
    float m = s1[0] / (float)M;
    mean[c] = m;
    var[c]  = s2[0] / (float)M - m * m;  // biased var, matches jnp.var
  }
}
__global__ void bn_apply(const float* __restrict__ X, float* Y,
                         const float* __restrict__ mean, const float* __restrict__ var,
                         const float* __restrict__ g, const float* __restrict__ b,
                         long M, int N, float slope /* <0: no act, else leaky */) {
  long i = (long)blockIdx.x * 256 + threadIdx.x;
  if (i >= M * N) return;
  int c = (int)(i % N);
  float v = (X[i] - mean[c]) * rsqrtf(var[c] + 1e-5f);
  if (g) v *= g[c];
  if (b) v += b[c];
  if (slope >= 0.f) v = v > 0.f ? v : slope * v;
  Y[i] = v;
}

// ---------------------------------------------------------------------------
// GAT kernels
// ---------------------------------------------------------------------------
__global__ void attn_scores(const float* __restrict__ Hm, const float* __restrict__ Att,
                            float* S, long Nn) {
  long i = (long)blockIdx.x * 256 + threadIdx.x;
  if (i >= Nn * HEADS) return;
  long n = i / HEADS; int h = (int)(i % HEADS);
  const float* hp = Hm + n * HID + h * CHD;
  const float* ap = Att + h * CHD;
  float s = 0.f;
  for (int c = 0; c < CHD; c++) s += hp[c] * ap[c];
  S[i] = s;
}
__global__ void edge_we_dot(const float* __restrict__ We, const float* __restrict__ Ae,
                            float* wdot) {
  int h = threadIdx.x;
  if (h >= HEADS) return;
  float s = 0.f;
  for (int c = 0; c < CHD; c++) s += We[h * CHD + c] * Ae[h * CHD + c];
  wdot[h] = s;
}
__global__ void edge_alpha_max(const int* __restrict__ src, const int* __restrict__ dst,
                               const float* __restrict__ attr,
                               const float* __restrict__ Ss, const float* __restrict__ Sd,
                               const float* __restrict__ wdot,
                               float* alpha, float* mx, long E) {
  long i = (long)blockIdx.x * 256 + threadIdx.x;
  if (i >= E * HEADS) return;
  long e = i / HEADS; int h = (int)(i % HEADS);
  float a = Ss[(long)src[e] * HEADS + h] + Sd[(long)dst[e] * HEADS + h]
          + attr[e] * wdot[h];
  a = a > 0.f ? a : 0.2f * a;            // leaky_relu(0.2)
  alpha[i] = a;
  atomicMaxF(&mx[(long)dst[e] * HEADS + h], a);
}
__global__ void edge_exp_sum(const int* __restrict__ dst, float* alpha,
                             const float* __restrict__ mx, float* sm, long E) {
  long i = (long)blockIdx.x * 256 + threadIdx.x;
  if (i >= E * HEADS) return;
  long e = i / HEADS; int h = (int)(i % HEADS);
  float w = expf(alpha[i] - mx[(long)dst[e] * HEADS + h]);
  alpha[i] = w;
  atomicAdd(&sm[(long)dst[e] * HEADS + h], w);
}
__global__ void edge_scatter(const int* __restrict__ src, const int* __restrict__ dst,
                             const float* __restrict__ alpha, const float* __restrict__ sm,
                             const float* __restrict__ Hs, float* out, long E) {
  long e = blockIdx.x;         // one block per edge, 256 channels
  int  c = threadIdx.x;
  int  h = c >> 6;
  int  s = src[e], d = dst[e];
  float coef = alpha[e * HEADS + h] / (sm[(long)d * HEADS + h] + 1e-16f);
  atomicAdd(&out[(long)d * HID + c], Hs[(long)s * HID + c] * coef);
}

// ---------------------------------------------------------------------------
// SAGE kernels (mean aggregation)
// ---------------------------------------------------------------------------
__global__ void sage_agg(const int* __restrict__ src, const int* __restrict__ dst,
                         const float* __restrict__ X, float* agg, float* cnt, long E) {
  long e = blockIdx.x;
  int  c = threadIdx.x;
  int  s = src[e], d = dst[e];
  atomicAdd(&agg[(long)d * HID + c], X[(long)s * HID + c]);
  if (c == 0) atomicAdd(&cnt[d], 1.f);
}
__global__ void sage_mean(float* agg, const float* __restrict__ cnt, long Nd) {
  long i = (long)blockIdx.x * 256 + threadIdx.x;
  if (i >= Nd * HID) return;
  agg[i] /= fmaxf(cnt[i / HID], 1.f);
}

// ---------------------------------------------------------------------------
// Host orchestration
// ---------------------------------------------------------------------------
extern "C" void kernel_launch(void* const* d_in, const int* in_sizes, int n_in,
                              void* d_out, int out_size, void* d_ws, size_t ws_size,
                              hipStream_t stream) {
  const int  kN0 = 40000, kN1 = 16000, kN2 = 4096, kF = 4096, kOUT = 40;
  const long kEF = 600000, kE0 = 240000, kE1 = 61440, kE2 = 120000;

  #define INF(i)  ((const float*)d_in[(i)])
  #define INI(i)  ((const int*)d_in[(i)])

  // ---- workspace carve (bump allocator, 256B aligned) ----
  char* w = (char*)d_ws;
  auto alloc = [&](size_t bytes) -> float* {
    float* p = (float*)w;
    w += (bytes + 255) & ~(size_t)255;
    return p;
  };
  float* bufX  = alloc((size_t)kN0 * 512 * 4);   // concat / generic scratch
  float* tp    = alloc((size_t)kN0 * HID * 4);   // temp_layer intermediate 1
  float* tq    = alloc((size_t)kN0 * HID * 4);   // temp_layer intermediate 2
  float* obs   = alloc((size_t)kN0 * HID * 4);   // current obs representation
  float* rep   = alloc((size_t)kN0 * HID * 4);   // GAT/SAGE secondary rep
  float* feat  = alloc((size_t)kF  * HID * 4);
  float* hsb   = alloc((size_t)kF  * HID * 4);   // GAT src transform (max 4096 rows)
  float* hdb   = alloc((size_t)kN0 * HID * 4);   // GAT dst transform (max 40000 rows)
  float* sAb   = alloc((size_t)kF  * HEADS * 4);
  float* sDb   = alloc((size_t)kN0 * HEADS * 4);
  float* mxb   = alloc((size_t)kN0 * HEADS * 4);
  float* smb   = alloc((size_t)kN0 * HEADS * 4);
  float* alphab= alloc((size_t)kEF * HEADS * 4);
  float* cntb  = alloc((size_t)kN1 * 4);
  float* meanb = alloc(HID * 4);
  float* varb  = alloc(HID * 4);
  float* wdotb = alloc(HEADS * 4);
  (void)ws_size; (void)n_in; (void)in_sizes; (void)out_size;

  auto gemm = [&](const float* A, int lda, const float* B, int ldb,
                  float* C, int ldc, const float* bias,
                  const float* addsrc, int ldadd, int M, int N, int K, int act) {
    dim3 g((N + BN - 1) / BN, (M + BM - 1) / BM);
    gemm_bf16_wmma<<<g, 256, 0, stream>>>(A, lda, B, ldb, C, ldc, bias,
                                          addsrc, ldadd, M, N, K, act);
  };
  auto nb = [](long n) { return (unsigned)((n + 255) / 256); };

  // temp_layer: Linear -> BN(affine) -> Linear -> ReLU   (params at pi..pi+5)
  auto temp_layer = [&](const float* x, int rows, int K, int pi, float* out) {
    gemm(x, K, INF(pi), HID, tp, HID, INF(pi + 1), nullptr, 0, rows, HID, K, 0);
    bn_stats<<<HID, 256, 0, stream>>>(tp, rows, HID, meanb, varb);
    bn_apply<<<nb((long)rows * HID), 256, 0, stream>>>(
        tp, tq, meanb, varb, INF(pi + 2), INF(pi + 3), rows, HID, -1.f);
    gemm(tq, HID, INF(pi + 4), HID, out, HID, INF(pi + 5), nullptr, 0, rows, HID, HID, 1);
  };

  // gat_conv(xs,xd,src,dst,attr) with params (Ws,Wd,As,Ad,We,Ae,b) at gi..gi+6
  auto gat = [&](const float* xs, int Ns, const float* xd, int Nd,
                 const int* src, const int* dst, const float* attr, long E,
                 int gi, float* out) {
    gemm(xs, HID, INF(gi + 0), HID, hsb, HID, nullptr, nullptr, 0, Ns, HID, HID, 0);
    gemm(xd, HID, INF(gi + 1), HID, hdb, HID, nullptr, nullptr, 0, Nd, HID, HID, 0);
    attn_scores<<<nb((long)Ns * HEADS), 256, 0, stream>>>(hsb, INF(gi + 2), sAb, Ns);
    attn_scores<<<nb((long)Nd * HEADS), 256, 0, stream>>>(hdb, INF(gi + 3), sDb, Nd);
    edge_we_dot<<<1, 32, 0, stream>>>(INF(gi + 4), INF(gi + 5), wdotb);
    fill_f32<<<nb((long)Nd * HEADS), 256, 0, stream>>>(mxb, -1e30f, (long)Nd * HEADS);
    fill_f32<<<nb((long)Nd * HEADS), 256, 0, stream>>>(smb, 0.f, (long)Nd * HEADS);
    edge_alpha_max<<<nb(E * HEADS), 256, 0, stream>>>(src, dst, attr, sAb, sDb,
                                                      wdotb, alphab, mxb, E);
    edge_exp_sum<<<nb(E * HEADS), 256, 0, stream>>>(dst, alphab, mxb, smb, E);
    fill_rows<<<nb((long)Nd * HID), 256, 0, stream>>>(out, INF(gi + 6), Nd, HID);
    edge_scatter<<<(unsigned)E, 256, 0, stream>>>(src, dst, alphab, smb, hsb, out, E);
  };

  // ---------------- pipeline ----------------
  // 1. obs = temp_layer(obs_features [N0,128], p_proj_obs0 @12)
  temp_layer(INF(0), kN0, 128, 12, obs);
  // 2. feat = temp_layer(feat_features [F,512], p_proj_feat0 @18)
  temp_layer(INF(1), kF, 512, 18, feat);
  // 3. g = gat(feat -> obs, fto_full, g_fto0 @48) -> rep[:N0]
  gat(feat, kF, obs, kN0, INI(2), INI(3), INF(4), kEF, 48, rep);
  // 4. obs = temp_layer(concat(obs, rep) [N0,512], p_obs_gat0 @24)
  concat2<<<nb((long)kN0 * 512), 256, 0, stream>>>(obs, rep, bufX, kN0, HID);
  temp_layer(bufX, kN0, 512, 24, obs);
  // 5. rep = temp_layer(obs, p_proj0 @30)
  temp_layer(obs, kN0, HID, 30, rep);
  // 6. sage0: xs=rep[N0], xd=rep[:N1], s0 @69  -> obs[:N1] (ELU)
  fill_f32<<<nb((long)kN1 * HID), 256, 0, stream>>>(bufX, 0.f, (long)kN1 * HID);
  fill_f32<<<nb((long)kN1), 256, 0, stream>>>(cntb, 0.f, (long)kN1);
  sage_agg<<<(unsigned)kE0, 256, 0, stream>>>(INI(5), INI(6), rep, bufX, cntb, kE0);
  sage_mean<<<nb((long)kN1 * HID), 256, 0, stream>>>(bufX, cntb, kN1);
  gemm(bufX, HID, INF(69), HID, tq, HID, INF(70), nullptr, 0, kN1, HID, HID, 0);
  gemm(rep, HID, INF(71), HID, obs, HID, nullptr, tq, HID, kN1, HID, HID, 2);
  // 7. rep = temp_layer(obs[:N1], p_proj1 @36)
  temp_layer(obs, kN1, HID, 36, rep);
  // 8. sage1: xs=rep[:N1], xd=rep[:N2], s1 @72 -> obs[:N2] (ELU)
  fill_f32<<<nb((long)kN2 * HID), 256, 0, stream>>>(bufX, 0.f, (long)kN2 * HID);
  fill_f32<<<nb((long)kN2), 256, 0, stream>>>(cntb, 0.f, (long)kN2);
  sage_agg<<<(unsigned)kE1, 256, 0, stream>>>(INI(7), INI(8), rep, bufX, cntb, kE1);
  sage_mean<<<nb((long)kN2 * HID), 256, 0, stream>>>(bufX, cntb, kN2);
  gemm(bufX, HID, INF(72), HID, tq, HID, INF(73), nullptr, 0, kN2, HID, HID, 0);
  gemm(rep, HID, INF(74), HID, obs, HID, nullptr, tq, HID, kN2, HID, HID, 2);
  // 9. g2 = gat(obs[:N2] -> feat, otf, g_otf0 @55) -> rep[:F]
  gat(obs, kN2, feat, kF, INI(9), INI(10), INF(11), kE2, 55, rep);
  // 10. feat = temp_layer(concat(feat, rep[:F]) [F,512], p_feat_gat0 @42)
  concat2<<<nb((long)kF * 512), 256, 0, stream>>>(feat, rep, bufX, kF, HID);
  temp_layer(bufX, kF, 512, 42, feat);
  // 11. g3 = gat(feat -> obs[:N2], flipped otf edges, g_repeat @62) -> rep[:N2]
  gat(feat, kF, obs, kN2, INI(10), INI(9), INF(11), kE2, 62, rep);
  // 12. obs = concat(obs[:N2], rep[:N2]) @ Wr(75) + br(76) -> tq[:N2]
  concat2<<<nb((long)kN2 * 512), 256, 0, stream>>>(obs, rep, bufX, kN2, HID);
  gemm(bufX, 512, INF(75), HID, tq, HID, INF(76), nullptr, 0, kN2, HID, 512, 0);
  // 13. project_Y: Linear -> BN(no affine) -> LeakyReLU(0.01) -> Linear
  gemm(tq, HID, INF(77), HID, tp, HID, INF(78), nullptr, 0, kN2, HID, HID, 0);
  bn_stats<<<HID, 256, 0, stream>>>(tp, kN2, HID, meanb, varb);
  bn_apply<<<nb((long)kN2 * HID), 256, 0, stream>>>(
      tp, bufX, meanb, varb, nullptr, nullptr, kN2, HID, 0.01f);
  gemm(bufX, HID, INF(79), kOUT, (float*)d_out, kOUT, INF(80), nullptr, 0,
       kN2, kOUT, HID, 0);
}
